// Phase3Stage2Block_4750233829427
// MI455X (gfx1250) — compile-verified, matrix operand order checked by reading, not run on hardware
//
#include <hip/hip_runtime.h>
#include <hip/hip_bf16.h>

// ---------------------------------------------------------------------------
// Sizes fixed by the reference: B=4, N=4096, D=512 -> BN=16384 tokens.
// ---------------------------------------------------------------------------
#define BN      16384
#define DDIM    512
#define KT_N    16      // number of K=32 chunks in 512
#define NT_N    32      // number of N=16 tiles in 512
#define DT_C    0.1f
#define EPS_C   1e-5f

typedef _Float16 h8   __attribute__((ext_vector_type(8)));
typedef _Float16 v16h __attribute__((ext_vector_type(16)));
typedef float    v8f  __attribute__((ext_vector_type(8)));

// Build a v16h A/B fragment from two 16-byte chunks. For A (row-major LDS,
// row stride 512 f16): chunk0 at k = kbase + 8*half, chunk1 at +16.
// For packed B the two chunks are contiguous (stride 8 f16).
__device__ __forceinline__ v16h frag_from(const _Float16* p0, const _Float16* p1) {
    union { v16h v; h8 h[2]; } u;
    u.h[0] = *(const h8*)p0;
    u.h[1] = *(const h8*)p1;
    return u.v;
}

__device__ __forceinline__ float wave_sum(float v) {
#pragma unroll
    for (int m = 16; m >= 1; m >>= 1) v += __shfl_xor(v, m, 32);
    return v;
}

// ---------------------------------------------------------------------------
// Pre-pack W1 (512x512 f32, row-major, W1[k][n]) into f16 WMMA B-fragment
// layout for 16x16x32: tile t = kt*32 + nt. Per tile, lane l (= N column
// l&15, K-half l>>4) holds 16 f16 contiguous:
//   dword v<4 : K = kt*32 + 8*(l>>4) + 2v   (+0,+1)
//   dword v>=4: K = kt*32 + 16 + 8*(l>>4) + 2(v-4)
// fwd: B[k][n] = W1[k][n];  bwd: B[k][n] = W1t[k][n] = W1[n][k]
// ---------------------------------------------------------------------------
__global__ void prepack_w1(const float* __restrict__ W1,
                           _Float16* __restrict__ fwd,
                           _Float16* __restrict__ bwd) {
    int idx = blockIdx.x * blockDim.x + threadIdx.x;  // 512 tiles * 32 lanes * 8 dwords
    if (idx >= 512 * 32 * 8) return;
    int v    = idx & 7;
    int lane = (idx >> 3) & 31;
    int t    = idx >> 8;
    int kt   = t >> 5;          // 0..15
    int nt   = t & 31;          // 0..31
    int half = lane >> 4;
    int n    = nt * 16 + (lane & 15);
    int koff = (v < 4) ? (8 * half + 2 * v) : (16 + 8 * half + 2 * (v - 4));
    int k0   = kt * 32 + koff;
    int o    = (t * 32 + lane) * 16 + 2 * v;
    fwd[o]     = (_Float16)W1[(size_t)k0 * DDIM + n];
    fwd[o + 1] = (_Float16)W1[(size_t)(k0 + 1) * DDIM + n];
    bwd[o]     = (_Float16)W1[(size_t)n * DDIM + k0];
    bwd[o + 1] = (_Float16)W1[(size_t)n * DDIM + k0 + 1];
}

// ---------------------------------------------------------------------------
// Imag path: out_i = LN2(LN1(z) + z), one wave32 per token (8 waves/block).
// ---------------------------------------------------------------------------
__global__ void imag_ln(const float* __restrict__ z,
                        const float* __restrict__ g1, const float* __restrict__ b1,
                        const float* __restrict__ g2, const float* __restrict__ b2,
                        float* __restrict__ out) {
    int wave = (blockIdx.x * blockDim.x + threadIdx.x) >> 5;
    int lane = threadIdx.x & 31;
    if (wave >= BN) return;
    const float* x = z + (size_t)wave * DDIM;
    float vals[16], s = 0.f, s2 = 0.f;
#pragma unroll
    for (int j = 0; j < 16; j++) {
        float v = x[j * 32 + lane];
        vals[j] = v; s += v; s2 += v * v;
    }
    s = wave_sum(s); s2 = wave_sum(s2);
    float mean = s * (1.f / DDIM);
    float rstd = rsqrtf(s2 * (1.f / DDIM) - mean * mean + EPS_C);
    float t[16], ts = 0.f, ts2 = 0.f;
#pragma unroll
    for (int j = 0; j < 16; j++) {
        int c = j * 32 + lane;
        float y  = g1[c] * (vals[j] - mean) * rstd + b1[c];
        float tt = y + vals[j];
        t[j] = tt; ts += tt; ts2 += tt * tt;
    }
    ts = wave_sum(ts); ts2 = wave_sum(ts2);
    float m2 = ts * (1.f / DDIM);
    float r2 = rsqrtf(ts2 * (1.f / DDIM) - m2 * m2 + EPS_C);
#pragma unroll
    for (int j = 0; j < 16; j++) {
        int c = j * 32 + lane;
        out[(size_t)wave * DDIM + c] = g2[c] * (t[j] - m2) * r2 + b2[c];
    }
}

// ---------------------------------------------------------------------------
// Real path mega-kernel. Block = 128 threads = 4 waves; each wave owns 32
// token rows (2 M-tiles) -> 128 rows per WG, grid = 128. LDS: qf16 + u
// (each 128x512 f16 = 128 KB) = 256 KB. Waves are independent: no barriers.
//
// Velocity-Verlet with 10 grad evals:
//   eval 0: g=grad(q0); p = -0.5*DT*g; q += DT*p
//   eval s (1..9): g=grad(q_s); p -= DT*g; q += DT*p    ==> q = q_10
// grad(q) = ((1 - tanh(qW1+b1)^2) * w2) @ W1^T
// ---------------------------------------------------------------------------
__global__ __launch_bounds__(128, 1) void hamiltonian_real(
    const float* __restrict__ zr,
    const float* __restrict__ ln1g, const float* __restrict__ ln1b,
    const float* __restrict__ ln2g, const float* __restrict__ ln2b,
    const _Float16* __restrict__ w1f, const _Float16* __restrict__ w1b,
    const float* __restrict__ b1v, const float* __restrict__ w2v,
    float* __restrict__ qws, float* __restrict__ pws,
    float* __restrict__ outr)
{
    extern __shared__ _Float16 smem[];
    _Float16* qf = smem;                    // [128][512] f16 operand copy of q
    _Float16* ub = smem + 128 * DDIM;       // [128][512] f16 backward A operand

    const int lane  = threadIdx.x & 31;
    const int wave  = threadIdx.x >> 5;     // 0..3
    const int rw    = wave * 32;            // wave row base within WG
    const int rowWG = blockIdx.x * 128;     // WG row base (global)
    const int half  = lane >> 4;
    const int nlo   = lane & 15;

    // ---------------- LN1 -> q0 (f32 ws + f16 LDS) ----------------
    for (int r = 0; r < 32; r++) {
        const int row = rw + r;
        const size_t g = (size_t)(rowWG + row) * DDIM;
        float vals[16], s = 0.f, s2 = 0.f;
#pragma unroll
        for (int j = 0; j < 16; j++) {
            float v = zr[g + j * 32 + lane];
            vals[j] = v; s += v; s2 += v * v;
        }
        s = wave_sum(s); s2 = wave_sum(s2);
        float mean = s * (1.f / DDIM);
        float rstd = rsqrtf(s2 * (1.f / DDIM) - mean * mean + EPS_C);
#pragma unroll
        for (int j = 0; j < 16; j++) {
            int c = j * 32 + lane;
            float q0 = ln1g[c] * (vals[j] - mean) * rstd + ln1b[c];
            qws[g + c] = q0;
            qf[row * DDIM + c] = (_Float16)q0;
        }
    }

    // ---------------- 10 grad evals (fused kick + drift) ----------------
    for (int s = 0; s < 10; s++) {
        // FORWARD: u = (1 - tanh(q@W1 + b1)^2) * w2
        for (int nt = 0; nt < NT_N; nt++) {
            v8f c0 = {}; v8f c1 = {};
            const _Float16* a0 = qf + (rw + nlo) * DDIM + 8 * half;
            const _Float16* a1 = a0 + 16 * DDIM;
            const _Float16* bp = w1f + ((size_t)nt * 32 + lane) * 16;
            __builtin_prefetch(bp, 0, 1);
#pragma unroll
            for (int kt = 0; kt < KT_N; kt++) {
                v16h bf  = frag_from(bp + (size_t)kt * 16384, bp + (size_t)kt * 16384 + 8);
                v16h af0 = frag_from(a0 + kt * 32, a0 + kt * 32 + 16);
                v16h af1 = frag_from(a1 + kt * 32, a1 + kt * 32 + 16);
                c0 = __builtin_amdgcn_wmma_f32_16x16x32_f16(false, af0, false, bf,
                                                            (short)0, c0, false, false);
                c1 = __builtin_amdgcn_wmma_f32_16x16x32_f16(false, af1, false, bf,
                                                            (short)0, c1, false, false);
            }
            const int n = nt * 16 + nlo;       // hidden index, lane-uniform per frag
            const float b1n = b1v[n];
            const float w2n = w2v[n];
#pragma unroll
            for (int e = 0; e < 8; e++) {
                float h0 = tanhf(c0[e] + b1n);
                ub[(rw + e + 8 * half) * DDIM + n]      = (_Float16)((1.f - h0 * h0) * w2n);
                float h1 = tanhf(c1[e] + b1n);
                ub[(rw + 16 + e + 8 * half) * DDIM + n] = (_Float16)((1.f - h1 * h1) * w2n);
            }
        }

        // BACKWARD: g = u @ W1^T, fused p-kick + q-drift + qf16 refresh
        const bool first = (s == 0);
        for (int nt = 0; nt < NT_N; nt++) {
            v8f c0 = {}; v8f c1 = {};
            const _Float16* a0 = ub + (rw + nlo) * DDIM + 8 * half;
            const _Float16* a1 = a0 + 16 * DDIM;
            const _Float16* bp = w1b + ((size_t)nt * 32 + lane) * 16;
            __builtin_prefetch(bp, 0, 1);
#pragma unroll
            for (int kt = 0; kt < KT_N; kt++) {
                v16h bf  = frag_from(bp + (size_t)kt * 16384, bp + (size_t)kt * 16384 + 8);
                v16h af0 = frag_from(a0 + kt * 32, a0 + kt * 32 + 16);
                v16h af1 = frag_from(a1 + kt * 32, a1 + kt * 32 + 16);
                c0 = __builtin_amdgcn_wmma_f32_16x16x32_f16(false, af0, false, bf,
                                                            (short)0, c0, false, false);
                c1 = __builtin_amdgcn_wmma_f32_16x16x32_f16(false, af1, false, bf,
                                                            (short)0, c1, false, false);
            }
            const int n = nt * 16 + nlo;       // output feature index
#pragma unroll
            for (int e = 0; e < 8; e++) {
                // M-tile 0
                {
                    int lrow = rw + e + 8 * half;
                    size_t gi = (size_t)(rowWG + lrow) * DDIM + n;
                    float pn = first ? (-0.5f * DT_C * c0[e]) : (pws[gi] - DT_C * c0[e]);
                    pws[gi] = pn;
                    float qn = qws[gi] + DT_C * pn;
                    qws[gi] = qn;
                    qf[lrow * DDIM + n] = (_Float16)qn;
                }
                // M-tile 1
                {
                    int lrow = rw + 16 + e + 8 * half;
                    size_t gi = (size_t)(rowWG + lrow) * DDIM + n;
                    float pn = first ? (-0.5f * DT_C * c1[e]) : (pws[gi] - DT_C * c1[e]);
                    pws[gi] = pn;
                    float qn = qws[gi] + DT_C * pn;
                    qws[gi] = qn;
                    qf[lrow * DDIM + n] = (_Float16)qn;
                }
            }
        }
    }

    // ---------------- out_r = LN2(q_final + z_real) ----------------
    for (int r = 0; r < 32; r++) {
        const int row = rw + r;
        const size_t g = (size_t)(rowWG + row) * DDIM;
        float xs[16], s = 0.f, s2 = 0.f;
#pragma unroll
        for (int j = 0; j < 16; j++) {
            int c = j * 32 + lane;
            float x = qws[g + c] + zr[g + c];
            xs[j] = x; s += x; s2 += x * x;
        }
        s = wave_sum(s); s2 = wave_sum(s2);
        float mean = s * (1.f / DDIM);
        float rstd = rsqrtf(s2 * (1.f / DDIM) - mean * mean + EPS_C);
#pragma unroll
        for (int j = 0; j < 16; j++) {
            int c = j * 32 + lane;
            outr[g + c] = ln2g[c] * (xs[j] - mean) * rstd + ln2b[c];
        }
    }
}

// ---------------------------------------------------------------------------
// Launch. Inputs in setup_inputs() order:
// 0 z_real 1 z_imag 2 ln1_gr 3 ln1_br 4 ln1_gi 5 ln1_bi
// 6 ln2_gr 7 ln2_br 8 ln2_gi 9 ln2_bi 10 W1 11 b1 12 W2 13 b2 (b2 unused)
// d_out = [out_r | out_i] flat f32.
// Workspace: qws (32MB) | pws (32MB) | w1f (512KB) | w1b (512KB) ~= 65MB.
// ---------------------------------------------------------------------------
extern "C" void kernel_launch(void* const* d_in, const int* in_sizes, int n_in,
                              void* d_out, int out_size, void* d_ws, size_t ws_size,
                              hipStream_t stream) {
    const float* z_real = (const float*)d_in[0];
    const float* z_imag = (const float*)d_in[1];
    const float* ln1_gr = (const float*)d_in[2];
    const float* ln1_br = (const float*)d_in[3];
    const float* ln1_gi = (const float*)d_in[4];
    const float* ln1_bi = (const float*)d_in[5];
    const float* ln2_gr = (const float*)d_in[6];
    const float* ln2_br = (const float*)d_in[7];
    const float* ln2_gi = (const float*)d_in[8];
    const float* ln2_bi = (const float*)d_in[9];
    const float* W1     = (const float*)d_in[10];
    const float* b1     = (const float*)d_in[11];
    const float* W2     = (const float*)d_in[12];   // (512,1) -> w2 column

    float* out_r = (float*)d_out;
    float* out_i = out_r + (size_t)BN * DDIM;

    float*    qws = (float*)d_ws;
    float*    pws = qws + (size_t)BN * DDIM;
    _Float16* w1f = (_Float16*)(pws + (size_t)BN * DDIM);
    _Float16* w1b = w1f + (size_t)DDIM * DDIM;

    // 1) Pack W1 and W1^T into WMMA B-fragment layout (f16).
    prepack_w1<<<512, 256, 0, stream>>>(W1, w1f, w1b);
    // 2) Imag path (independent, elementwise per token).
    imag_ln<<<BN / 8, 256, 0, stream>>>(z_imag, ln1_gi, ln1_bi, ln2_gi, ln2_bi, out_i);
    // 3) Real path: LN1 + 10-grad-eval leapfrog (WMMA) + residual + LN2.
    hamiltonian_real<<<128, 128, 256 * 1024, stream>>>(
        z_real, ln1_gr, ln1_br, ln2_gr, ln2_br,
        w1f, w1b, b1, W2, qws, pws, out_r);
}
